// DirectedMultiheadAttention_33054068310213
// MI455X (gfx1250) — matrix operand order repaired and checked by reading, not compiled, not run on hardware
//
#include <hip/hip_runtime.h>
#include <hip/hip_bf16.h>

#define T_DIM 2048
#define C_DIM 1024
#define N_B   2
#define N_H   16
#define H_D   64

typedef __bf16 bf16_t;
typedef __attribute__((ext_vector_type(8)))  __bf16 bf16x8;
typedef __attribute__((ext_vector_type(16))) __bf16 bf16x16;
typedef __attribute__((ext_vector_type(8)))  float  v8f;
typedef __attribute__((ext_vector_type(4)))  unsigned int v4u;

// exact pointee type expected by the async-to-LDS builtin (per hipcc diag)
typedef __attribute__((__vector_size__(4 * sizeof(int)))) int v4i;
typedef __attribute__((address_space(1))) v4i gv4i;  // global v4i
typedef __attribute__((address_space(3))) v4i lv4i;  // LDS v4i

#if defined(__AMDGCN__) && __has_builtin(__builtin_amdgcn_global_load_async_to_lds_b128)
#define HAVE_ASYNC_LDS 1
#else
#define HAVE_ASYNC_LDS 0
#endif

#if HAVE_ASYNC_LDS
#if __has_builtin(__builtin_amdgcn_s_wait_asynccnt)
#define WAIT_ASYNC(n) __builtin_amdgcn_s_wait_asynccnt(n)
#else
#define WAIT_ASYNC(n) asm volatile("s_wait_asynccnt " #n ::: "memory")
#endif
#else
#define WAIT_ASYNC(n)
#endif

// copy 16B global -> LDS (async on CDNA5, tracked by ASYNCcnt)
static __device__ __forceinline__ void cp16_lds(const bf16_t* g, bf16_t* l) {
#if HAVE_ASYNC_LDS
  __builtin_amdgcn_global_load_async_to_lds_b128(
      (gv4i*)(unsigned long long)g,
      (lv4i*)(unsigned int)(unsigned long long)l, 0, 0);
#else
  *reinterpret_cast<v4u*>(l) = *reinterpret_cast<const v4u*>(g);
#endif
}

// D = A(16x32 bf16) x B(32x16 bf16) + C(16x16 f32)
static __device__ __forceinline__ v8f wmma_bf16(bf16x16 a, bf16x16 b, v8f c) {
  return __builtin_amdgcn_wmma_f32_16x16x32_bf16(false, a, false, b, (short)0, c,
                                                 false, false);
}

// A-fragment (16x32 row-major): lane row M=lane%16; elems 0..15 -> K=16*hb+0..15
// (hb = lane/16)
static __device__ __forceinline__ bf16x16 load_a_frag(const bf16_t* p, int hb) {
  bf16x8 lo = *reinterpret_cast<const bf16x8*>(p + 16 * hb);
  bf16x8 hi = *reinterpret_cast<const bf16x8*>(p + 16 * hb + 8);
  return __builtin_shufflevector(lo, hi, 0, 1, 2, 3, 4, 5, 6, 7,
                                 8, 9, 10, 11, 12, 13, 14, 15);
}
// B-fragment (32x16): lane column N=lane%16; K = 16*hb + 0..15 (contiguous 32B)
static __device__ __forceinline__ bf16x16 load_b_frag(const bf16_t* p) {
  return *reinterpret_cast<const bf16x16*>(p);
}

static __device__ __forceinline__ float redmax16(float x) {
  x = fmaxf(x, __shfl_xor(x, 1, 16));
  x = fmaxf(x, __shfl_xor(x, 2, 16));
  x = fmaxf(x, __shfl_xor(x, 4, 16));
  x = fmaxf(x, __shfl_xor(x, 8, 16));
  return x;
}
static __device__ __forceinline__ float redsum16(float x) {
  x += __shfl_xor(x, 1, 16);
  x += __shfl_xor(x, 2, 16);
  x += __shfl_xor(x, 4, 16);
  x += __shfl_xor(x, 8, 16);
  return x;
}

// ---------------------------------------------------------------- convert
__global__ void cvt_f32_bf16_kernel(const float* __restrict__ src,
                                    bf16_t* __restrict__ dst, int n) {
  for (int i = blockIdx.x * blockDim.x + threadIdx.x; i < n;
       i += gridDim.x * blockDim.x)
    dst[i] = (bf16_t)src[i];
}

// ---------------------------------------------------------------- QKV GEMM
// grid: (T/64, C/64 == H, 3*B)  block: 128 (4 waves share the LDS weight tile)
__global__ __launch_bounds__(128) void qkv_gemm_kernel(
    const bf16_t* __restrict__ Xb, const bf16_t* __restrict__ Wq,
    const bf16_t* __restrict__ Wk, const bf16_t* __restrict__ Wv,
    const float* __restrict__ cv, const float* __restrict__ cb,
    bf16_t* __restrict__ Qb, bf16_t* __restrict__ Kb,
    bf16_t* __restrict__ Vt) {
  __shared__ __align__(16) bf16_t Bt[2][64][32];  // double-buffered W tile

  const int tid = threadIdx.x;
  const int lane = tid & 31;
  const int wave = tid >> 5;
  const int hb = lane >> 4;
  const int l16 = lane & 15;
  const int mat = blockIdx.z / N_B;  // 0=Q 1=K 2=V
  const int b = blockIdx.z % N_B;
  const int h = blockIdx.y;          // C/64 == N_H
  const int t0 = blockIdx.x * 64 + wave * 16;

  const bf16_t* W = (mat == 0) ? Wq : (mat == 1) ? Wk : Wv;
  const bf16_t* arow = Xb + (size_t)(b * T_DIM + t0 + l16) * C_DIM;

  // per-thread staging chunks: 256 x 16B chunks cover the 64x32 tile
  const int c0i = tid, c1i = tid + 128;
  auto stage = [&](int k0, int buf) {
    cp16_lds(W + (size_t)(h * 64 + (c0i >> 2)) * C_DIM + k0 + (c0i & 3) * 8,
             &Bt[buf][c0i >> 2][(c0i & 3) * 8]);
    cp16_lds(W + (size_t)(h * 64 + (c1i >> 2)) * C_DIM + k0 + (c1i & 3) * 8,
             &Bt[buf][c1i >> 2][(c1i & 3) * 8]);
  };

  v8f acc[4] = {v8f{}, v8f{}, v8f{}, v8f{}};
  stage(0, 0);
  bf16x16 a_cur = load_a_frag(arow, hb);

  for (int k0 = 0; k0 < C_DIM; k0 += 32) {
    const int cur = (k0 >> 5) & 1;
    const bool more = (k0 + 32) < C_DIM;
    if (more) {
      stage(k0 + 32, cur ^ 1);
      WAIT_ASYNC(2);
    } else {
      WAIT_ASYNC(0);
    }
    __syncthreads();
    __builtin_prefetch(arow + k0 + 128, 0, 3);
    bf16x16 a_next = more ? load_a_frag(arow + k0 + 32, hb) : a_cur;
#pragma unroll
    for (int nt = 0; nt < 4; ++nt)
      acc[nt] = wmma_bf16(a_cur, load_b_frag(&Bt[cur][nt * 16 + l16][16 * hb]),
                          acc[nt]);
    a_cur = a_next;
    __syncthreads();  // all waves done with buffer `cur` before it is restaged
  }

  const int bh = b * N_H + h;
  if (mat == 2) {
    // Vt[(bh*64 + d)*T + t]: lane holds fixed d, 8 consecutive t -> packed b128
#pragma unroll
    for (int nt = 0; nt < 4; ++nt) {
      const int d = nt * 16 + l16;
      bf16x8 pk;
#pragma unroll
      for (int v = 0; v < 8; ++v) pk[v] = (bf16_t)acc[nt][v];
      *reinterpret_cast<bf16x8*>(Vt + (size_t)(bh * 64 + d) * T_DIM + t0 +
                                 8 * hb) = pk;
    }
  } else {
    bf16_t* dst = (mat == 0) ? Qb : Kb;
#pragma unroll
    for (int nt = 0; nt < 4; ++nt) {
      const int n = h * 64 + nt * 16 + l16;
#pragma unroll
      for (int v = 0; v < 8; ++v) {
        const int t = t0 + v + 8 * hb;
        float val = acc[nt][v];
        if (mat == 0)
          val = (val + cb[n] * cv[(size_t)(b * T_DIM + t) * C_DIM + n]) * 0.125f;
        dst[(size_t)(bh * T_DIM + t) * H_D + nt * 16 + l16] = (bf16_t)val;
      }
    }
  }
}

// ---------------------------------------------------------------- attention
// grid: (B*H, T/64)  block: 128; K/V tiles staged once per block in LDS
__global__ __launch_bounds__(128) void attn_kernel(
    const bf16_t* __restrict__ Qb, const bf16_t* __restrict__ Kb,
    const bf16_t* __restrict__ Vt, bf16_t* __restrict__ AO) {
  __shared__ __align__(16) bf16_t Ks[2][32][64];   // k-tile (t x d)
  __shared__ __align__(16) bf16_t Vs[2][64][32];   // v-tile (d x t)
  __shared__ __align__(32) bf16_t Pl[4][16][32];   // per-wave private P slice

  const int tid = threadIdx.x;
  const int lane = tid & 31;
  const int wave = tid >> 5;
  const int hb = lane >> 4;
  const int l16 = lane & 15;
  const int bh = blockIdx.x;
  const int b = bh / N_H;
  const int h = bh % N_H;
  const int q0 = blockIdx.y * 64 + wave * 16;

  const bf16_t* kbase = Kb + (size_t)bh * T_DIM * H_D;
  const bf16_t* vbase = Vt + (size_t)bh * H_D * T_DIM;

  const int c0i = tid, c1i = tid + 128;
  auto stage = [&](int c0, int buf) {
    // K tile: 32 rows(t) x 64(d); chunk c -> row c>>3, off (c&7)*8
    cp16_lds(kbase + (size_t)(c0 + (c0i >> 3)) * H_D + (c0i & 7) * 8,
             &Ks[buf][c0i >> 3][(c0i & 7) * 8]);
    cp16_lds(kbase + (size_t)(c0 + (c1i >> 3)) * H_D + (c1i & 7) * 8,
             &Ks[buf][c1i >> 3][(c1i & 7) * 8]);
    // V tile: 64 rows(d) x 32(t); chunk c -> row c>>2, off (c&3)*8
    cp16_lds(vbase + (size_t)(c0i >> 2) * T_DIM + c0 + (c0i & 3) * 8,
             &Vs[buf][c0i >> 2][(c0i & 3) * 8]);
    cp16_lds(vbase + (size_t)(c1i >> 2) * T_DIM + c0 + (c1i & 3) * 8,
             &Vs[buf][c1i >> 2][(c1i & 3) * 8]);
  };

  // Q fragments (scale 1/sqrt(hd) folded in at projection time)
  const bf16_t* qrow = Qb + (size_t)(bh * T_DIM + q0 + l16) * H_D;
  const bf16x16 qf0 = load_a_frag(qrow, hb);
  const bf16x16 qf1 = load_a_frag(qrow + 32, hb);

  v8f o[4] = {v8f{}, v8f{}, v8f{}, v8f{}};
  float m[8], lsum[8];
#pragma unroll
  for (int v = 0; v < 8; ++v) {
    m[v] = -1e30f;
    lsum[v] = 0.0f;
  }

  // block-wide k extent (barriers must stay uniform across the 4 waves)
  const int nkt = (blockIdx.y * 64 + 95) >> 5;
  stage(0, 0);

  for (int kt = 0; kt < nkt; ++kt) {
    const int c0 = kt * 32;
    const int cur = kt & 1;
    const bool more = (kt + 1) < nkt;
    if (more) {
      stage(c0 + 32, cur ^ 1);
      WAIT_ASYNC(4);
    } else {
      WAIT_ASYNC(0);
    }
    __syncthreads();

    if (c0 <= q0 + 15) {  // this wave still has unmasked columns in this tile
      v8f s0 = {}, s1 = {};
      s0 = wmma_bf16(qf0, load_b_frag(&Ks[cur][l16][16 * hb]), s0);
      s0 = wmma_bf16(qf1, load_b_frag(&Ks[cur][l16][32 + 16 * hb]), s0);
      s1 = wmma_bf16(qf0, load_b_frag(&Ks[cur][16 + l16][16 * hb]), s1);
      s1 = wmma_bf16(qf1, load_b_frag(&Ks[cur][16 + l16][32 + 16 * hb]), s1);

      // causal mask + online softmax (row v+8*hb lives in VGPR v per half)
#pragma unroll
      for (int v = 0; v < 8; ++v) {
        const int row = q0 + v + 8 * hb;
        const float sa = (c0 + l16 <= row) ? s0[v] : -1e30f;
        const float sb = (c0 + 16 + l16 <= row) ? s1[v] : -1e30f;
        const float mx = redmax16(fmaxf(sa, sb));
        const float mn = fmaxf(m[v], mx);
        const float alpha = __expf(m[v] - mn);
        const float pa = __expf(sa - mn);
        const float pb = __expf(sb - mn);
        lsum[v] = lsum[v] * alpha + redsum16(pa + pb);
        m[v] = mn;
#pragma unroll
        for (int nt = 0; nt < 4; ++nt) o[nt][v] *= alpha;
        Pl[wave][v + 8 * hb][l16] = (bf16_t)pa;
        Pl[wave][v + 8 * hb][16 + l16] = (bf16_t)pb;
      }
      // D-layout -> A-layout through our private LDS slice (same-wave DS order)
      const bf16x16 pf = load_a_frag(&Pl[wave][l16][0], hb);
#pragma unroll
      for (int nt = 0; nt < 4; ++nt)
        o[nt] = wmma_bf16(pf, load_b_frag(&Vs[cur][nt * 16 + l16][16 * hb]),
                          o[nt]);
    }
    __syncthreads();  // everyone done with buffer `cur` before restage
  }

  float inv[8];
#pragma unroll
  for (int v = 0; v < 8; ++v) inv[v] = 1.0f / lsum[v];
#pragma unroll
  for (int nt = 0; nt < 4; ++nt) {
#pragma unroll
    for (int v = 0; v < 8; ++v) {
      const int t = q0 + v + 8 * hb;
      AO[(size_t)(b * T_DIM + t) * C_DIM + h * 64 + nt * 16 + l16] =
          (bf16_t)(o[nt][v] * inv[v]);
    }
  }
}

// ---------------------------------------------------------------- out proj
// grid: (T/64, C/64, B)  block: 128
__global__ __launch_bounds__(128) void out_gemm_kernel(
    const bf16_t* __restrict__ AO, const bf16_t* __restrict__ Wo,
    const float* __restrict__ bo, float* __restrict__ out) {
  __shared__ __align__(16) bf16_t Bt[2][64][32];

  const int tid = threadIdx.x;
  const int lane = tid & 31;
  const int wave = tid >> 5;
  const int hb = lane >> 4;
  const int l16 = lane & 15;
  const int b = blockIdx.z;
  const int n0 = blockIdx.y * 64;
  const int t0 = blockIdx.x * 64 + wave * 16;

  const bf16_t* arow = AO + (size_t)(b * T_DIM + t0 + l16) * C_DIM;
  const int c0i = tid, c1i = tid + 128;
  auto stage = [&](int k0, int buf) {
    cp16_lds(Wo + (size_t)(n0 + (c0i >> 2)) * C_DIM + k0 + (c0i & 3) * 8,
             &Bt[buf][c0i >> 2][(c0i & 3) * 8]);
    cp16_lds(Wo + (size_t)(n0 + (c1i >> 2)) * C_DIM + k0 + (c1i & 3) * 8,
             &Bt[buf][c1i >> 2][(c1i & 3) * 8]);
  };

  v8f acc[4] = {v8f{}, v8f{}, v8f{}, v8f{}};
  stage(0, 0);
  bf16x16 a_cur = load_a_frag(arow, hb);

  for (int k0 = 0; k0 < C_DIM; k0 += 32) {
    const int cur = (k0 >> 5) & 1;
    const bool more = (k0 + 32) < C_DIM;
    if (more) {
      stage(k0 + 32, cur ^ 1);
      WAIT_ASYNC(2);
    } else {
      WAIT_ASYNC(0);
    }
    __syncthreads();
    __builtin_prefetch(arow + k0 + 128, 0, 3);
    bf16x16 a_next = more ? load_a_frag(arow + k0 + 32, hb) : a_cur;
#pragma unroll
    for (int nt = 0; nt < 4; ++nt)
      acc[nt] = wmma_bf16(a_cur, load_b_frag(&Bt[cur][nt * 16 + l16][16 * hb]),
                          acc[nt]);
    a_cur = a_next;
    __syncthreads();
  }

#pragma unroll
  for (int nt = 0; nt < 4; ++nt) {
    const int n = n0 + nt * 16 + l16;
    const float bias = bo[n];
#pragma unroll
    for (int v = 0; v < 8; ++v) {
      const int t = t0 + v + 8 * hb;
      out[(size_t)(b * T_DIM + t) * C_DIM + n] = acc[nt][v] + bias;
    }
  }
}

extern "C" void kernel_launch(void* const* d_in, const int* in_sizes, int n_in,
                              void* d_out, int out_size, void* d_ws,
                              size_t ws_size, hipStream_t stream) {
  (void)in_sizes; (void)n_in; (void)out_size; (void)ws_size;
  const float* x = (const float*)d_in[0];
  // d_in[1] = mask: causal tril, handled analytically
  const float* cv = (const float*)d_in[2];
  const float* wq = (const float*)d_in[3];
  const float* wk = (const float*)d_in[4];
  const float* wv = (const float*)d_in[5];
  const float* wo = (const float*)d_in[6];
  const float* bo = (const float*)d_in[7];
  const float* cb = (const float*)d_in[8];
  float* out = (float*)d_out;

  const size_t nX = (size_t)N_B * T_DIM * C_DIM;  // 4,194,304
  const size_t nW = (size_t)C_DIM * C_DIM;        // 1,048,576

  bf16_t* Xb = (bf16_t*)d_ws;
  bf16_t* Wqb = Xb + nX;
  bf16_t* Wkb = Wqb + nW;
  bf16_t* Wvb = Wkb + nW;
  bf16_t* Wob = Wvb + nW;
  bf16_t* Qb = Wob + nW;
  bf16_t* Kb = Qb + nX;
  bf16_t* Vt = Kb + nX;
  bf16_t* AO = Vt + nX;  // ~48 MiB of bf16 staging total

  cvt_f32_bf16_kernel<<<4096, 256, 0, stream>>>(x, Xb, (int)nX);
  cvt_f32_bf16_kernel<<<2048, 256, 0, stream>>>(wq, Wqb, (int)nW);
  cvt_f32_bf16_kernel<<<2048, 256, 0, stream>>>(wk, Wkb, (int)nW);
  cvt_f32_bf16_kernel<<<2048, 256, 0, stream>>>(wv, Wvb, (int)nW);
  cvt_f32_bf16_kernel<<<2048, 256, 0, stream>>>(wo, Wob, (int)nW);

  qkv_gemm_kernel<<<dim3(T_DIM / 64, C_DIM / 64, 3 * N_B), 128, 0, stream>>>(
      Xb, Wqb, Wkb, Wvb, cv, cb, Qb, Kb, Vt);

  attn_kernel<<<dim3(N_B * N_H, T_DIM / 64), 128, 0, stream>>>(Qb, Kb, Vt, AO);

  out_gemm_kernel<<<dim3(T_DIM / 64, C_DIM / 64, N_B), 128, 0, stream>>>(
      AO, Wob, bo, out);
}
